// InteractionNet_13340168421980
// MI455X (gfx1250) — matrix operand order repaired and checked by reading, not compiled, not run on hardware
//
#include <hip/hip_runtime.h>

#define F_DIM 128
#define NB 20
#define PITCH 136   // 128 + 8 halves of padding to avoid LDS bank conflicts
#define WMAT (F_DIM * PITCH)          // halves per pre-transposed weight image
#define LN_EPS 1e-5f

typedef __attribute__((ext_vector_type(16))) _Float16 v16h;
typedef __attribute__((ext_vector_type(8)))  _Float16 v8h;
typedef __attribute__((ext_vector_type(8)))  float    v8f;

// ---------------------------------------------------------------------------
// WMMA helpers (gfx1250 wave32, V_WMMA_F32_16X16X32_F16)
// ---------------------------------------------------------------------------
__device__ __forceinline__ v8f wmma16(v16h a, v16h b, v8f c) {
    return __builtin_amdgcn_wmma_f32_16x16x32_f16(
        false, a, false, b, (short)0, c, false, false);
}

// A fragment: 16x32 f16 tile from row-major LDS stage [16][PITCH].
// lane = (khalf<<4)|m ; elements 0..7 -> K = kc*32 + kh*8 + i,
// elements 8..15 -> K = kc*32 + 16 + kh*8 + i.
__device__ __forceinline__ v16h frag_A(const _Float16* s, int lane, int kc) {
    const int m  = lane & 15;
    const int kh = lane >> 4;
    const _Float16* r = s + m * PITCH + kc * 32 + kh * 8;
    v8h lo = *(const v8h*)(r);
    v8h hi = *(const v8h*)(r + 16);
    v16h a;
#pragma unroll
    for (int i = 0; i < 8; ++i) { a[i] = lo[i]; a[8 + i] = hi[i]; }
    return a;
}

// B fragment: 32x16 f16 tile from transposed weight image WT[n][k] in LDS.
// lane holds column n = t*16+(lane&15); kh = lane>>4; elements 0..7 -> K = kc*32 + kh*8 + i,
// elements 8..15 -> K = kc*32 + 16 + kh*8 + i.
__device__ __forceinline__ v16h frag_B(const _Float16* wt, int lane, int t, int kc) {
    const int n  = t * 16 + (lane & 15);
    const int kh = lane >> 4;
    const _Float16* r = wt + n * PITCH + kc * 32 + kh * 8;
    v8h lo = *(const v8h*)(r);
    v8h hi = *(const v8h*)(r + 16);
    v16h b;
#pragma unroll
    for (int i = 0; i < 8; ++i) { b[i] = lo[i]; b[8 + i] = hi[i]; }
    return b;
}

__device__ __forceinline__ float silu(float x) {
    return x / (1.0f + __expf(-x));
}

// ---------------------------------------------------------------------------
// K0a: init accumulators (workspace is poisoned 0xAA by the harness)
// ---------------------------------------------------------------------------
__global__ void k_init(const float* __restrict__ atom,
                       float* __restrict__ atom_acc,
                       float* __restrict__ agg1,
                       float* __restrict__ disp_acc, int n) {
    int idx = blockIdx.x * blockDim.x + threadIdx.x;
    int n3f = n * 3 * F_DIM;
    if (idx < n3f) { agg1[idx] = 0.0f; disp_acc[idx] = 0.0f; }
    if (idx < n * F_DIM) atom_acc[idx] = atom[idx];
}

// ---------------------------------------------------------------------------
// K0b: pre-convert one [128,128] f32 weight matrix into a transposed,
// pitched f16 image WT[n*PITCH + k] ready for direct LDS staging.
// ---------------------------------------------------------------------------
__global__ void k_wconv(const float* __restrict__ W, _Float16* __restrict__ out) {
    int idx = blockIdx.x * blockDim.x + threadIdx.x;   // 16384 threads
    int k = idx >> 7, n = idx & 127;
    out[n * PITCH + k] = (_Float16)W[idx];
}

// ---------------------------------------------------------------------------
// Node mlp2: out = silu(x*W1+b1)*W2+b2 with pre-converted f16 weight images.
// 256 threads = 8 waves; each wave owns 16 rows x 128 cols.
// ---------------------------------------------------------------------------
__global__ void __launch_bounds__(256)
k_mlp2_node(const float* __restrict__ x,
            const _Float16* __restrict__ wt1, const float* __restrict__ b1,
            const _Float16* __restrict__ wt2, const float* __restrict__ b2,
            float* __restrict__ out, int M) {
    __shared__ __align__(16) _Float16 sWT1[WMAT];
    __shared__ __align__(16) _Float16 sWT2[WMAT];
    __shared__ __align__(16) _Float16 sStage[8 * 16 * PITCH];
    __shared__ float sB1[F_DIM], sB2[F_DIM];

    const int lane = threadIdx.x & 31;
    const int wave = threadIdx.x >> 5;

    // Straight 16B copies of the pre-transposed weight images.
    for (int i = threadIdx.x; i < WMAT / 8; i += blockDim.x) {
        ((v8h*)sWT1)[i] = ((const v8h*)wt1)[i];
        ((v8h*)sWT2)[i] = ((const v8h*)wt2)[i];
    }
    for (int i = threadIdx.x; i < F_DIM; i += blockDim.x) {
        sB1[i] = b1 ? b1[i] : 0.0f;
        sB2[i] = b2 ? b2[i] : 0.0f;
    }

    _Float16* stg = sStage + wave * 16 * PITCH;
    const int row0 = blockIdx.x * 128 + wave * 16;

    for (int i = lane; i < 16 * F_DIM; i += 32) {
        int rr = i >> 7, cc = i & 127;
        int gr = row0 + rr;
        float v = (gr < M) ? x[(size_t)gr * F_DIM + cc] : 0.0f;
        stg[rr * PITCH + cc] = (_Float16)v;
    }
    __syncthreads();

    v16h a0 = frag_A(stg, lane, 0), a1 = frag_A(stg, lane, 1),
         a2 = frag_A(stg, lane, 2), a3 = frag_A(stg, lane, 3);
    const int mb = (lane >> 4) * 8;
#pragma unroll
    for (int t = 0; t < 8; ++t) {
        const int n = t * 16 + (lane & 15);
        const float bb = sB1[n];
        v8f c = { bb, bb, bb, bb, bb, bb, bb, bb };
        c = wmma16(a0, frag_B(sWT1, lane, t, 0), c);
        c = wmma16(a1, frag_B(sWT1, lane, t, 1), c);
        c = wmma16(a2, frag_B(sWT1, lane, t, 2), c);
        c = wmma16(a3, frag_B(sWT1, lane, t, 3), c);
#pragma unroll
        for (int g = 0; g < 8; ++g)
            stg[(mb + g) * PITCH + n] = (_Float16)silu(c[g]);
    }

    v16h h0 = frag_A(stg, lane, 0), h1 = frag_A(stg, lane, 1),
         h2 = frag_A(stg, lane, 2), h3 = frag_A(stg, lane, 3);
#pragma unroll
    for (int t = 0; t < 8; ++t) {
        const int n = t * 16 + (lane & 15);
        const float bb = sB2[n];
        v8f c = { bb, bb, bb, bb, bb, bb, bb, bb };
        c = wmma16(h0, frag_B(sWT2, lane, t, 0), c);
        c = wmma16(h1, frag_B(sWT2, lane, t, 1), c);
        c = wmma16(h2, frag_B(sWT2, lane, t, 2), c);
        c = wmma16(h3, frag_B(sWT2, lane, t, 3), c);
#pragma unroll
        for (int g = 0; g < 8; ++g) {
            int gr = row0 + mb + g;
            if (gr < M) out[(size_t)gr * F_DIM + n] = c[g];
        }
    }
}

// ---------------------------------------------------------------------------
// K2: inv_message = (dist*empW + b) * h[src] * h[dst];  stored as f16.
//     atom_acc += seg_sum(inv_message)   (one block of 128 per edge)
// ---------------------------------------------------------------------------
__global__ void __launch_bounds__(128)
k_edge_inv(const float* __restrict__ dist, const float* __restrict__ empW,
           const float* __restrict__ empb, const float* __restrict__ h,
           const int* __restrict__ src, const int* __restrict__ dst,
           _Float16* __restrict__ msg, float* __restrict__ atom_acc, int E) {
    const int e = blockIdx.x;
    const int f = threadIdx.x;
    __shared__ float sd[NB];
    if (f < NB) sd[f] = dist[(size_t)e * NB + f];
    __syncthreads();
    const int s = src[e], d = dst[e];
    float acc = empb[f];
#pragma unroll
    for (int j = 0; j < NB; ++j) acc = fmaf(sd[j], empW[j * F_DIM + f], acc);
    float m = acc * h[(size_t)s * F_DIM + f] * h[(size_t)d * F_DIM + f];
    msg[(size_t)e * F_DIM + f] = (_Float16)m;
    atomicAdd(&atom_acc[s * F_DIM + f], m);
}

// ---------------------------------------------------------------------------
// One full mlp2 pass (layer1 + silu + layer2) over a wave's staged A
// fragments, followed by the equivariant scatter (eq1 or eq2 flavor).
// ---------------------------------------------------------------------------
template <bool USE_DISP_E>
__device__ __forceinline__ void mlp_scatter_pass(
    const _Float16* sW1, const float* sB1, const _Float16* sW2, const float* sB2,
    _Float16* stg, v16h a0, v16h a1, v16h a2, v16h a3,
    int lane, int wave, int e0b, int E,
    const int* sSrc, const int* sDst, const float (*sDe)[3],
    const float* __restrict__ dispn, float* __restrict__ acc_out) {
    const int mb = (lane >> 4) * 8;
#pragma unroll
    for (int t = 0; t < 8; ++t) {
        const int n = t * 16 + (lane & 15);
        const float bb = sB1[n];
        v8f c = { bb, bb, bb, bb, bb, bb, bb, bb };
        c = wmma16(a0, frag_B(sW1, lane, t, 0), c);
        c = wmma16(a1, frag_B(sW1, lane, t, 1), c);
        c = wmma16(a2, frag_B(sW1, lane, t, 2), c);
        c = wmma16(a3, frag_B(sW1, lane, t, 3), c);
#pragma unroll
        for (int g = 0; g < 8; ++g)
            stg[(mb + g) * PITCH + n] = (_Float16)silu(c[g]);
    }

    v16h h0 = frag_A(stg, lane, 0), h1 = frag_A(stg, lane, 1),
         h2 = frag_A(stg, lane, 2), h3 = frag_A(stg, lane, 3);
#pragma unroll
    for (int t = 0; t < 8; ++t) {
        const int n = t * 16 + (lane & 15);
        const float bb = sB2[n];
        v8f c = { bb, bb, bb, bb, bb, bb, bb, bb };
        c = wmma16(h0, frag_B(sW2, lane, t, 0), c);
        c = wmma16(h1, frag_B(sW2, lane, t, 1), c);
        c = wmma16(h2, frag_B(sW2, lane, t, 2), c);
        c = wmma16(h3, frag_B(sW2, lane, t, 3), c);
#pragma unroll
        for (int g = 0; g < 8; ++g) {
            const int li = wave * 16 + mb + g;     // local edge within block
            const int e  = e0b + li;
            if (e >= E) continue;
            const int s  = sSrc[li];
            const float val = c[g];
            if (USE_DISP_E) {
#pragma unroll
                for (int d = 0; d < 3; ++d)
                    atomicAdd(&acc_out[((size_t)s * 3 + d) * F_DIM + n],
                              val * sDe[li][d]);
            } else {
                const int dd = sDst[li];
#pragma unroll
                for (int d = 0; d < 3; ++d)
                    atomicAdd(&acc_out[((size_t)s * 3 + d) * F_DIM + n],
                              val * dispn[((size_t)dd * 3 + d) * F_DIM + n]);
            }
        }
    }
}

// ---------------------------------------------------------------------------
// K3: fused eq1 + eq2 edge MLPs. Stages the f16 inv_message tile once,
// reuses the A fragments for both MLP pipelines, scatters both results.
//   eq1: agg1     += seg(mlp2(msg,eq1) * disp_edge)
//   eq2: disp_acc += seg(mlp2(msg,eq2) * disp_node[dst])
// ---------------------------------------------------------------------------
__global__ void __launch_bounds__(256)
k_edge_mlp_fused(const _Float16* __restrict__ msg,
                 const _Float16* __restrict__ w11, const float* __restrict__ b11,
                 const _Float16* __restrict__ w12, const float* __restrict__ b12,
                 const _Float16* __restrict__ w21, const float* __restrict__ b21,
                 const _Float16* __restrict__ w22, const float* __restrict__ b22,
                 const int* __restrict__ src, const int* __restrict__ dst,
                 const float* __restrict__ dispe,   // [E,3]
                 const float* __restrict__ dispn,   // [N,3,F]
                 float* __restrict__ agg1, float* __restrict__ disp_acc, int E) {
    __shared__ __align__(16) _Float16 sW11[WMAT];
    __shared__ __align__(16) _Float16 sW12[WMAT];
    __shared__ __align__(16) _Float16 sW21[WMAT];
    __shared__ __align__(16) _Float16 sW22[WMAT];
    __shared__ __align__(16) _Float16 sStage[8 * 16 * PITCH];
    __shared__ float sB11[F_DIM], sB12[F_DIM], sB21[F_DIM], sB22[F_DIM];
    __shared__ int   sSrc[128], sDst[128];
    __shared__ float sDe[128][3];

    const int lane = threadIdx.x & 31;
    const int wave = threadIdx.x >> 5;
    const int e0b  = blockIdx.x * 128;

    for (int i = threadIdx.x; i < WMAT / 8; i += blockDim.x) {
        ((v8h*)sW11)[i] = ((const v8h*)w11)[i];
        ((v8h*)sW12)[i] = ((const v8h*)w12)[i];
        ((v8h*)sW21)[i] = ((const v8h*)w21)[i];
        ((v8h*)sW22)[i] = ((const v8h*)w22)[i];
    }
    for (int i = threadIdx.x; i < F_DIM; i += blockDim.x) {
        sB11[i] = b11[i]; sB12[i] = b12[i];
        sB21[i] = b21[i]; sB22[i] = b22[i];
    }
    for (int i = threadIdx.x; i < 128; i += blockDim.x) {
        int e = e0b + i;
        sSrc[i] = (e < E) ? src[e] : 0;
        sDst[i] = (e < E) ? dst[e] : 0;
#pragma unroll
        for (int d = 0; d < 3; ++d)
            sDe[i][d] = (e < E) ? dispe[(size_t)e * 3 + d] : 0.0f;
    }

    _Float16* stg = sStage + wave * 16 * PITCH;
    const int row0 = e0b + wave * 16;

    // Hint the next block's message tile toward the caches.
    if ((size_t)(row0 + 128) < (size_t)E)
        __builtin_prefetch(&msg[(size_t)(row0 + 128) * F_DIM], 0, 1);

    // Stage this wave's 16 f16 message rows with straight 16B copies.
    for (int i = lane; i < 16 * (F_DIM / 8); i += 32) {
        int rr = i >> 4, ck = i & 15;
        int gr = row0 + rr;
        v8h v = { 0, 0, 0, 0, 0, 0, 0, 0 };
        if (gr < E) v = *(const v8h*)(msg + (size_t)gr * F_DIM + ck * 8);
        *(v8h*)(stg + rr * PITCH + ck * 8) = v;
    }
    __syncthreads();

    v16h a0 = frag_A(stg, lane, 0), a1 = frag_A(stg, lane, 1),
         a2 = frag_A(stg, lane, 2), a3 = frag_A(stg, lane, 3);

    // eq1 pass (scales by disp_edge, scatters into agg1)
    mlp_scatter_pass<true>(sW11, sB11, sW12, sB12, stg, a0, a1, a2, a3,
                           lane, wave, e0b, E, sSrc, sDst, sDe, dispn, agg1);
    // eq2 pass (scales by gathered disp_node[dst], scatters into disp_acc)
    mlp_scatter_pass<false>(sW21, sB21, sW22, sB22, stg, a0, a1, a2, a3,
                            lane, wave, e0b, E, sSrc, sDst, sDe, dispn, disp_acc);
}

// ---------------------------------------------------------------------------
// K5: eq3 = eq3_inv[dst] * agg1[dst] scattered into disp_acc over src.
// ---------------------------------------------------------------------------
__global__ void __launch_bounds__(128)
k_eq3_scatter(const float* __restrict__ eq3inv, const float* __restrict__ agg1,
              const int* __restrict__ src, const int* __restrict__ dst,
              float* __restrict__ disp_acc, int E) {
    const int e = blockIdx.x / 3;
    const int d = blockIdx.x % 3;
    const int f = threadIdx.x;
    const int s = src[e], t = dst[e];
    float v = eq3inv[(size_t)t * F_DIM + f] *
              agg1[((size_t)t * 3 + d) * F_DIM + f];
    atomicAdd(&disp_acc[((size_t)s * 3 + d) * F_DIM + f], v);
}

// ---------------------------------------------------------------------------
// K6b: final coupling + LayerNorm; also writes force/disp outputs.
// ---------------------------------------------------------------------------
__global__ void __launch_bounds__(128)
k_finalize(const float* __restrict__ atom_acc, const float* __restrict__ u,
           const float* __restrict__ force_in, const float* __restrict__ agg1,
           const float* __restrict__ disp_in, const float* __restrict__ disp_acc,
           const float* __restrict__ ln_g, const float* __restrict__ ln_b,
           float* __restrict__ out_atom, float* __restrict__ out_force,
           float* __restrict__ out_disp, int n) {
    const int i = blockIdx.x;
    const int f = threadIdx.x;
    __shared__ float red[F_DIM];

    float s = 0.0f;
#pragma unroll
    for (int d = 0; d < 3; ++d) {
        size_t idx = ((size_t)i * 3 + d) * F_DIM + f;
        float fo = force_in[idx] + agg1[idx];
        float dp = disp_in[idx] + disp_acc[idx];
        out_force[idx] = fo;
        out_disp[idx]  = dp;
        s -= fo * dp;
    }
    size_t nf = (size_t)i * F_DIM + f;
    float x = atom_acc[nf] + u[nf] * s;

    red[f] = x;
    __syncthreads();
    for (int off = 64; off > 0; off >>= 1) {
        if (f < off) red[f] += red[f + off];
        __syncthreads();
    }
    float mu = red[0] * (1.0f / F_DIM);
    __syncthreads();
    float dxm = x - mu;
    red[f] = dxm * dxm;
    __syncthreads();
    for (int off = 64; off > 0; off >>= 1) {
        if (f < off) red[f] += red[f + off];
        __syncthreads();
    }
    float var = red[0] * (1.0f / F_DIM);
    out_atom[nf] = dxm * rsqrtf(var + LN_EPS) * ln_g[f] + ln_b[f];
}

// ---------------------------------------------------------------------------
// Host side
// ---------------------------------------------------------------------------
extern "C" void kernel_launch(void* const* d_in, const int* in_sizes, int n_in,
                              void* d_out, int out_size, void* d_ws, size_t ws_size,
                              hipStream_t stream) {
    const float* atom   = (const float*)d_in[0];
    const float* force  = (const float*)d_in[1];
    const float* disp   = (const float*)d_in[2];
    const float* dispe  = (const float*)d_in[3];
    const float* dist   = (const float*)d_in[4];
    const int*   ei     = (const int*)d_in[5];
    const float* Wf32[10] = {
        (const float*)d_in[6],  (const float*)d_in[8],    // nmp W1, W2
        (const float*)d_in[12], (const float*)d_in[14],   // eq1 W1, W2
        (const float*)d_in[16], (const float*)d_in[18],   // eq2 W1, W2
        (const float*)d_in[20], (const float*)d_in[21],   // eq3 W1, W2
        (const float*)d_in[22], (const float*)d_in[24]    // upd W1, W2
    };
    const float* nmpb1 = (const float*)d_in[7];
    const float* nmpb2 = (const float*)d_in[9];
    const float* empW  = (const float*)d_in[10];
    const float* empb  = (const float*)d_in[11];
    const float* eq1b1 = (const float*)d_in[13];
    const float* eq1b2 = (const float*)d_in[15];
    const float* eq2b1 = (const float*)d_in[17];
    const float* eq2b2 = (const float*)d_in[19];
    const float* updb1 = (const float*)d_in[23];
    const float* updb2 = (const float*)d_in[25];
    const float* lng   = (const float*)d_in[26];
    const float* lnb   = (const float*)d_in[27];

    const int n = in_sizes[0] / F_DIM;
    const int E = in_sizes[5] / 2;
    const int* src = ei;
    const int* dst = ei + E;

    // --- workspace carve (256B aligned slices) ---
    char* wsb = (char*)d_ws;
    auto carve = [&](size_t bytes) {
        void* p = (void*)wsb;
        wsb += (bytes + 255) & ~(size_t)255;
        return p;
    };
    float*     h        = (float*)carve((size_t)n * F_DIM * 4);
    float*     atom_acc = (float*)carve((size_t)n * F_DIM * 4);
    float*     agg1     = (float*)carve((size_t)n * 3 * F_DIM * 4);
    float*     disp_acc = (float*)carve((size_t)n * 3 * F_DIM * 4);
    float*     eq3i     = (float*)carve((size_t)n * F_DIM * 4);
    float*     u        = (float*)carve((size_t)n * F_DIM * 4);
    _Float16*  msg      = (_Float16*)carve((size_t)E * F_DIM * 2);
    _Float16*  Wh[10];
    for (int i = 0; i < 10; ++i) Wh[i] = (_Float16*)carve((size_t)WMAT * 2);

    float* out_atom  = (float*)d_out;
    float* out_force = out_atom + (size_t)n * F_DIM;
    float* out_disp  = out_force + (size_t)n * 3 * F_DIM;

    const int nodeBlocks = (n + 127) / 128;
    const int edgeBlocks = (E + 127) / 128;

    // Pre-convert all weight matrices to transposed/pitched f16 images.
    for (int i = 0; i < 10; ++i)
        k_wconv<<<(F_DIM * F_DIM) / 256, 256, 0, stream>>>(Wf32[i], Wh[i]);

    k_init<<<(n * 3 * F_DIM + 255) / 256, 256, 0, stream>>>(
        atom, atom_acc, agg1, disp_acc, n);

    // h = mlp2(atom_node, nmp)
    k_mlp2_node<<<nodeBlocks, 256, 0, stream>>>(
        atom, Wh[0], nmpb1, Wh[1], nmpb2, h, n);

    // inv_message (f16) + atom accumulation
    k_edge_inv<<<E, 128, 0, stream>>>(
        dist, empW, empb, h, src, dst, msg, atom_acc, E);

    // fused eq1 + eq2 edge MLPs with scatter
    k_edge_mlp_fused<<<edgeBlocks, 256, 0, stream>>>(
        msg, Wh[2], eq1b1, Wh[3], eq1b2, Wh[4], eq2b1, Wh[5], eq2b2,
        src, dst, dispe, disp, agg1, disp_acc, E);

    // eq3_inv = mlp2_nobias(atom_acc, eq3)
    k_mlp2_node<<<nodeBlocks, 256, 0, stream>>>(
        atom_acc, Wh[6], nullptr, Wh[7], nullptr, eq3i, n);

    // disp_acc += seg(eq3_inv[dst] * agg1[dst])
    k_eq3_scatter<<<E * 3, 128, 0, stream>>>(eq3i, agg1, src, dst, disp_acc, E);

    // u = mlp2(atom_acc, upd)
    k_mlp2_node<<<nodeBlocks, 256, 0, stream>>>(
        atom_acc, Wh[8], updb1, Wh[9], updb2, u, n);

    k_finalize<<<n, 128, 0, stream>>>(
        atom_acc, u, force, agg1, disp, disp_acc, lng, lnb,
        out_atom, out_force, out_disp, n);
}
